// Decoder_77309411757
// MI455X (gfx1250) — matrix operand order repaired and checked by reading, not compiled, not run on hardware
//
#include <hip/hip_runtime.h>
#include <stdint.h>

typedef __attribute__((ext_vector_type(16))) __bf16 v16bf;
typedef __attribute__((ext_vector_type(8)))  float  v8f;
typedef __attribute__((ext_vector_type(4)))  float  v4f;

#define B_   32
#define LX_  64
#define T_   128
#define TS   127            // decode steps (T-1)
#define D_   512
#define DW_  512
#define V_   32000
#define GN   2048           // 4*D
#define LIN  2048           // LSTM_IN = 2*DW + 2*D
#define MALL (TS*B_)        // 4064

__device__ __forceinline__ float sigmoidf_(float x){ return 1.0f/(1.0f+__expf(-x)); }

// ---- WMMA fragment loaders: f32 global -> bf16 fragments (wave32, contiguous k per lane half, same for A and B) ----

// A-matrix 16x32 bf16 tile: lane L -> row (L&15); elements 0..15 = K[k0..k0+15],
// where k0 = ktile + (L>=16 ? 16 : 0).
__device__ __forceinline__ v16bf load_a_frag(const float* A, int lda, int m, int k0){
    const v4f* p4 = (const v4f*)(A + (long long)m*lda + k0);
    float t[16];
    *(v4f*)(t+0)  = p4[0];
    *(v4f*)(t+4)  = p4[1];
    *(v4f*)(t+8)  = p4[2];
    *(v4f*)(t+12) = p4[3];
    v16bf f;
#pragma unroll
    for (int i=0;i<16;i++) f[i] = (__bf16)t[i];
    return f;
}

// B-matrix 32x16 bf16 tile from row-major W(N,K): lane L -> col (L&15);
// elements 0..15 = W[n][k0..k0+15], k0 = ktile + (L>=16 ? 16 : 0).
__device__ __forceinline__ v16bf load_b_frag(const float* Bw, int ldb, int n, int k0){
    const v4f* p4 = (const v4f*)(Bw + (long long)n*ldb + k0);
    float t[16];
    *(v4f*)(t+0)  = p4[0];
    *(v4f*)(t+4)  = p4[1];
    *(v4f*)(t+8)  = p4[2];
    *(v4f*)(t+12) = p4[3];
    v16bf f;
#pragma unroll
    for (int i=0;i<16;i++) f[i] = (__bf16)t[i];
    return f;
}

// C[m][n] = act( Cinit[m][n] + sum_k A[m][k]*Bw[n][k] + bias[n] )
// One wave computes a 32x32 tile: 2 A-frags + 2 B-frags -> 4 WMMAs per k-step.
// Requires M%32==0, N%32==0, K%32==0 (true for every GEMM in this model).
__global__ void wmma_gemm_kernel(const float* __restrict__ A, int lda,
                                 const float* __restrict__ Bw, int ldb,
                                 float* C, long long ldc,
                                 const float* Cinit, long long ldci,
                                 const float* __restrict__ bias,
                                 int M, int N, int K, int act)
{
    const int lane = threadIdx.x & 31;
    const int wid  = threadIdx.x >> 5;
    const int nN = N >> 5;
    const int nM = M >> 5;
    const int gw = blockIdx.x * (blockDim.x >> 5) + wid;
    if (gw >= nM*nN) return;
    const int tn = gw % nN, tm = gw / nN;
    const int mbase = tm << 5, nbase = tn << 5;
    const int half = lane >> 4, lr = lane & 15;
    const int n0 = nbase + lr;
    const int n1 = n0 + 16;
    const int ak = half << 4;   // A k-offset: +0 / +16
    const int bk = half << 4;   // B k-offset: +0 / +16

    v8f acc00 = {}, acc01 = {}, acc10 = {}, acc11 = {};
    if (Cinit) {
#pragma unroll
        for (int v=0; v<8; v++) {
            int r = mbase + (half<<3) + v;
            acc00[v] = Cinit[(long long)r*ldci + n0];
            acc01[v] = Cinit[(long long)r*ldci + n1];
            acc10[v] = Cinit[(long long)(r+16)*ldci + n0];
            acc11[v] = Cinit[(long long)(r+16)*ldci + n1];
        }
    }

    for (int k = 0; k < K; k += 32) {
        if (k + 32 < K) {
            __builtin_prefetch(Bw + (long long)n0*ldb + k + bk + 32, 0, 1); // global_prefetch_b8
            __builtin_prefetch(Bw + (long long)n1*ldb + k + bk + 32, 0, 1);
        }
        v16bf b0 = load_b_frag(Bw, ldb, n0, k + bk);
        v16bf b1 = load_b_frag(Bw, ldb, n1, k + bk);
        v16bf a0 = load_a_frag(A, lda, mbase + lr,      k + ak);
        v16bf a1 = load_a_frag(A, lda, mbase + 16 + lr, k + ak);
        acc00 = __builtin_amdgcn_wmma_f32_16x16x32_bf16(false, a0, false, b0, (short)0, acc00, false, false);
        acc01 = __builtin_amdgcn_wmma_f32_16x16x32_bf16(false, a0, false, b1, (short)0, acc01, false, false);
        acc10 = __builtin_amdgcn_wmma_f32_16x16x32_bf16(false, a1, false, b0, (short)0, acc10, false, false);
        acc11 = __builtin_amdgcn_wmma_f32_16x16x32_bf16(false, a1, false, b1, (short)0, acc11, false, false);
    }

    const float bv0 = bias ? bias[n0] : 0.0f;
    const float bv1 = bias ? bias[n1] : 0.0f;
#pragma unroll
    for (int v=0; v<8; v++) {
        int r = mbase + (half<<3) + v;
        float x00 = acc00[v] + bv0;
        float x01 = acc01[v] + bv1;
        float x10 = acc10[v] + bv0;
        float x11 = acc11[v] + bv1;
        if (act) { x00 = tanhf(x00); x01 = tanhf(x01); x10 = tanhf(x10); x11 = tanhf(x11); }
        C[(long long)r*ldc + n0]      = x00;
        C[(long long)r*ldc + n1]      = x01;
        C[(long long)(r+16)*ldc + n0] = x10;
        C[(long long)(r+16)*ldc + n1] = x11;
    }
}

// ---- Small helper kernels ----

__global__ void gather_x_emb_kernel(const int* __restrict__ x_train,
                                    const float* __restrict__ word_emb,
                                    float* __restrict__ xg){
    int idx = blockIdx.x*blockDim.x + threadIdx.x;
    if (idx >= MALL*D_) return;
    int m = idx >> 9, d = idx & 511;      // row m = t*32 + b
    int t = m >> 5, b = m & 31;
    int tok = x_train[b*T_ + t];
    xg[idx] = word_emb[(long long)tok*DW_ + d];
}

__global__ void attr_kernel(const int* __restrict__ y_train,
                            const float* __restrict__ tab,
                            float* __restrict__ attr){
    int idx = blockIdx.x*blockDim.x + threadIdx.x;
    if (idx >= B_*DW_) return;
    int b = idx >> 9, d = idx & 511;
    attr[idx] = tab[(long long)y_train[b*2]  *DW_ + d]
              + tab[(long long)y_train[b*2+1]*DW_ + d];
}

// base_gates[m][n] += attr_gates[m%32][n] + b_ih[n] + b_hh[n]
__global__ void add_base_kernel(float* __restrict__ bg,
                                const float* __restrict__ ag,
                                const float* __restrict__ b_ih,
                                const float* __restrict__ b_hh){
    int idx = blockIdx.x*blockDim.x + threadIdx.x;
    if (idx >= MALL*GN) return;
    int n = idx & 2047;
    int b = (idx >> 11) & 31;
    bg[idx] += ag[b*GN + n] + b_ih[n] + b_hh[n];
}

__global__ void zero_kernel(float* __restrict__ p, int n){
    int idx = blockIdx.x*blockDim.x + threadIdx.x;
    if (idx < n) p[idx] = 0.0f;
}

// LSTM cell: gates (32x2048) [i|f|g|o], update c,h; mirror h into hc[:,0:512]
__global__ void lstm_cell_kernel(const float* __restrict__ gates,
                                 float* __restrict__ c, float* __restrict__ h,
                                 float* __restrict__ hc){
    int idx = blockIdx.x*blockDim.x + threadIdx.x;
    if (idx >= B_*D_) return;
    int b = idx >> 9, d = idx & 511;
    const float* g = gates + b*GN;
    float gi = g[d], gf = g[512+d], gg = g[1024+d], go = g[1536+d];
    float cn = sigmoidf_(gf)*c[idx] + sigmoidf_(gi)*tanhf(gg);
    float hn = sigmoidf_(go)*tanhf(cn);
    c[idx] = cn;
    h[idx] = hn;
    hc[b*(3*D_) + d] = hn;
}

// scores[b][l] = sum_d tanh(x_enc_k[b][l][d] + q[b][d]) * W_att[d] + b_att
// one wave32 per (b,l); lane handles 16 contiguous d, then shuffle-reduce.
__global__ void att_scores_kernel(const float* __restrict__ xk,
                                  const float* __restrict__ q,
                                  const float* __restrict__ W_att,
                                  const float* __restrict__ b_att,
                                  float* __restrict__ scores){
    int gw = (blockIdx.x*blockDim.x + threadIdx.x) >> 5;
    int lane = threadIdx.x & 31;
    if (gw >= B_*LX_) return;
    int b = gw >> 6;
    const float* xp = xk + (long long)gw*D_;
    const float* qp = q + b*D_;
    float s = 0.0f;
#pragma unroll 4
    for (int i=0;i<16;i++) {
        int d = lane*16 + i;
        s += tanhf(xp[d] + qp[d]) * W_att[d];
    }
#pragma unroll
    for (int off=16; off; off>>=1) s += __shfl_xor(s, off, 32);
    if (lane == 0) scores[gw] = s + b_att[0];
}

// per-batch softmax over 64 scores, then ctx[b][d] = sum_l w[l]*x_enc[b][l][d];
// ctx feeds both `feed` (next LSTM input) and hc[:,512:1536] (for W_cr GEMM).
__global__ void softmax_ctx_kernel(const float* __restrict__ scores,
                                   const float* __restrict__ x_enc,
                                   float* __restrict__ feed,
                                   float* __restrict__ hc){
    __shared__ float sm[LX_];
    int b = blockIdx.x;
    int tid = threadIdx.x;
    if (tid < LX_) sm[tid] = scores[b*LX_ + tid];
    __syncthreads();
    if (tid == 0) {
        float mx = sm[0];
        for (int l=1;l<LX_;l++) mx = fmaxf(mx, sm[l]);
        float s = 0.0f;
        for (int l=0;l<LX_;l++){ float e = __expf(sm[l]-mx); sm[l]=e; s+=e; }
        float inv = 1.0f/s;
        for (int l=0;l<LX_;l++) sm[l] *= inv;
    }
    __syncthreads();
    for (int d = tid; d < 2*D_; d += blockDim.x) {
        const float* xp = x_enc + (long long)b*LX_*(2*D_) + d;
        float acc = 0.0f;
#pragma unroll 8
        for (int l=0;l<LX_;l++) acc += sm[l]*xp[l*(2*D_)];
        feed[b*(2*D_) + d] = acc;
        hc[b*(3*D_) + D_ + d] = acc;
    }
}

extern "C" void kernel_launch(void* const* d_in, const int* in_sizes, int n_in,
                              void* d_out, int out_size, void* d_ws, size_t ws_size,
                              hipStream_t stream) {
    const float* x_enc    = (const float*)d_in[0];
    const float* x_enc_k  = (const float*)d_in[1];
    const float* dec_h0   = (const float*)d_in[2];
    const float* dec_c0   = (const float*)d_in[3];
    /* d_in[4] x_mask: unused by reference */
    const int*   y_train  = (const int*)d_in[5];
    const int*   x_train  = (const int*)d_in[6];
    const float* word_emb = (const float*)d_in[7];
    const float* attr_tab = (const float*)d_in[8];
    const float* W_ih     = (const float*)d_in[9];
    const float* W_hh     = (const float*)d_in[10];
    const float* b_ih     = (const float*)d_in[11];
    const float* b_hh     = (const float*)d_in[12];
    const float* W_trg    = (const float*)d_in[13];
    const float* b_trg    = (const float*)d_in[14];
    const float* W_att    = (const float*)d_in[15];
    const float* b_att    = (const float*)d_in[16];
    const float* W_cr     = (const float*)d_in[17];
    const float* W_ro     = (const float*)d_in[18];
    float* out = (float*)d_out;

    char* wp = (char*)d_ws;
    auto alloc = [&](size_t elems){ float* p = (float*)wp; wp += elems*sizeof(float); return p; };
    float* x_embg     = alloc((size_t)MALL*D_);   // (t*32+b, 512) gathered embeddings
    float* attr       = alloc((size_t)B_*DW_);
    float* base_gates = alloc((size_t)MALL*GN);   // x@W_ihx^T + attr@W_iha^T + biases
    float* attr_gates = alloc((size_t)B_*GN);
    float* gates      = alloc((size_t)B_*GN);
    float* h          = alloc((size_t)B_*D_);
    float* c          = alloc((size_t)B_*D_);
    float* feed       = alloc((size_t)B_*2*D_);
    float* hc         = alloc((size_t)B_*3*D_);   // [h | ctx]
    float* q          = alloc((size_t)B_*D_);
    float* scores     = alloc((size_t)B_*LX_);
    float* pres       = alloc((size_t)MALL*D_);   // stored (b,t,d) for direct logits rows

    auto gemm = [&](const float* A, int lda, const float* Bw, int ldb,
                    float* C, long long ldc, const float* Cinit, long long ldci,
                    const float* bias, int M, int N, int K, int act){
        int waves  = (M >> 5) * (N >> 5);
        int blocks = (waves + 7) / 8;
        wmma_gemm_kernel<<<blocks, 256, 0, stream>>>(A, lda, Bw, ldb, C, ldc,
                                                     Cinit, ldci, bias, M, N, K, act);
    };

    // ---- Precompute (hoisted out of the recurrence) ----
    gather_x_emb_kernel<<<(MALL*D_+255)/256, 256, 0, stream>>>(x_train, word_emb, x_embg);
    attr_kernel<<<(B_*DW_+255)/256, 256, 0, stream>>>(y_train, attr_tab, attr);
    // base_gates = x_emb @ W_ih[:,0:512]^T        (4064 x 2048 x 512)
    gemm(x_embg, D_, W_ih, LIN, base_gates, GN, nullptr, 0, nullptr, MALL, GN, D_, 0);
    // attr_gates = attr @ W_ih[:,512:1024]^T      (32 x 2048 x 512)
    gemm(attr, DW_, W_ih + DW_, LIN, attr_gates, GN, nullptr, 0, nullptr, B_, GN, DW_, 0);
    add_base_kernel<<<(MALL*GN+255)/256, 256, 0, stream>>>(base_gates, attr_gates, b_ih, b_hh);

    hipMemcpyAsync(h, dec_h0, (size_t)B_*D_*sizeof(float), hipMemcpyDeviceToDevice, stream);
    hipMemcpyAsync(c, dec_c0, (size_t)B_*D_*sizeof(float), hipMemcpyDeviceToDevice, stream);
    zero_kernel<<<(B_*2*D_+255)/256, 256, 0, stream>>>(feed, B_*2*D_);

    // ---- Recurrence ----
    for (int t = 0; t < TS; t++) {
        // gates = base_gates[t] + feed @ W_ih[:,1024:2048]^T
        gemm(feed, 2*D_, W_ih + 2*DW_, LIN, gates, GN,
             base_gates + (size_t)t*B_*GN, GN, nullptr, B_, GN, 2*D_, 0);
        // gates += h @ W_hh^T
        gemm(h, D_, W_hh, D_, gates, GN, gates, GN, nullptr, B_, GN, D_, 0);
        lstm_cell_kernel<<<(B_*D_+255)/256, 256, 0, stream>>>(gates, c, h, hc);
        // q = h @ W_trg^T + b_trg
        gemm(h, D_, W_trg, D_, q, D_, nullptr, 0, b_trg, B_, D_, D_, 0);
        att_scores_kernel<<<(B_*LX_*32+255)/256, 256, 0, stream>>>(x_enc_k, q, W_att, b_att, scores);
        softmax_ctx_kernel<<<B_, 256, 0, stream>>>(scores, x_enc, feed, hc);
        // pres[b][t] = tanh(hc @ W_cr^T); row b of the 32-row tile lands at (b*127 + t)
        gemm(hc, 3*D_, W_cr, 3*D_, pres + (size_t)t*D_, (long long)TS*D_,
             nullptr, 0, nullptr, B_, D_, 3*D_, 1);
    }

    // ---- logits (b*127+t, v) = pres @ W_ro^T : 4064 x 32000 x 512, output-BW bound ----
    gemm(pres, D_, W_ro, D_, out, V_, nullptr, 0, nullptr, MALL, V_, D_, 0);
}
